// CommonModule_42838003810964
// MI455X (gfx1250) — matrix operand order repaired and checked by reading, not compiled, not run on hardware
//
#include <hip/hip_runtime.h>
#include <hip/hip_bf16.h>

// MHA: qv,kv,vv [B,S,V,H] f32 ; ov_w [D,D] f32 ; ov_b [D] f32 ; out [B,S,D] f32
// D = V*H, att_flat index d = b*H + h  (feature-major, then head)

#define B_ 2
#define S_ 2048
#define V_ 64
#define H_ 16
#define D_ 1024   // V_*H_

typedef __attribute__((ext_vector_type(16))) _Float16 v16h;
typedef __attribute__((ext_vector_type(8)))  _Float16 v8h;
typedef __attribute__((ext_vector_type(8)))  float    v8f;

// ---- WMMA fragment index maps (CDNA5 ISA 7.12.2, wave32) ----
// A 16x32 f16: lane L row M=L%16; element e -> K = {8g..8g+7} for e<8,
//              {16+8g..16+8g+7} for e>=8  (two contiguous 8-half runs).
// B 32x16 f16: lane L col N=L%16; same K map as A; this code loads A and B both with K = 16g+e.
// C/D 16x16 f32: VGPR r, lane L -> (M = r + 8*(L/16), N = L%16).

__device__ __forceinline__ v16h ld16c(const _Float16* p) {            // contiguous 16
    v8h lo = *(const v8h*)p;
    v8h hi = *(const v8h*)(p + 8);
    return __builtin_shufflevector(lo, hi, 0, 1, 2, 3, 4, 5, 6, 7,
                                   8, 9, 10, 11, 12, 13, 14, 15);
}
__device__ __forceinline__ v16h ld16s(const _Float16* p, int g) {     // A-map split
    v8h lo = *(const v8h*)(p + 8 * g);
    v8h hi = *(const v8h*)(p + 16 + 8 * g);
    return __builtin_shufflevector(lo, hi, 0, 1, 2, 3, 4, 5, 6, 7,
                                   8, 9, 10, 11, 12, 13, 14, 15);
}

__device__ __forceinline__ v8f wmma_f16(v16h a, v16h b, v8f c) {
    return __builtin_amdgcn_wmma_f32_16x16x32_f16(false, a, false, b, (short)0, c,
                                                  false, false);
}

__device__ __forceinline__ float redmax16(float v) {
    v = fmaxf(v, __shfl_xor(v, 1, 32));
    v = fmaxf(v, __shfl_xor(v, 2, 32));
    v = fmaxf(v, __shfl_xor(v, 4, 32));
    v = fmaxf(v, __shfl_xor(v, 8, 32));
    return v;
}
__device__ __forceinline__ float redsum16(float v) {
    v += __shfl_xor(v, 1, 32);
    v += __shfl_xor(v, 2, 32);
    v += __shfl_xor(v, 4, 32);
    v += __shfl_xor(v, 8, 32);
    return v;
}

// ---------------------------------------------------------------------------
// Packing kernels (one-time f32 -> f16 layout conversion; L2-resident reads,
// fully coalesced writes).
// ---------------------------------------------------------------------------
// dst[((a*H+h)*S + i)*V + b] = src[((a*S+i)*V + b)*H + h] * scale
__global__ __launch_bounds__(256)
void pack_headmajor_kernel(const float* __restrict__ src, _Float16* __restrict__ dst,
                           float scale) {
    size_t idx = (size_t)blockIdx.x * 256 + threadIdx.x;   // over B*H*S*V
    int b = (int)(idx % V_); size_t t = idx / V_;
    int i = (int)(t % S_);   t /= S_;
    int h = (int)(t % H_);
    int a = (int)(t / H_);
    dst[idx] = (_Float16)(src[(((size_t)a * S_ + i) * V_ + b) * H_ + h] * scale);
}
// dst[((a*H+h)*V + b)*S + j] = src[((a*S+j)*V + b)*H + h]   (V transposed)
__global__ __launch_bounds__(256)
void pack_v_transposed_kernel(const float* __restrict__ src,
                              _Float16* __restrict__ dst) {
    size_t idx = (size_t)blockIdx.x * 256 + threadIdx.x;   // over B*H*V*S
    int j = (int)(idx % S_); size_t t = idx / S_;
    int b = (int)(t % V_);   t /= V_;
    int h = (int)(t % H_);
    int a = (int)(t / H_);
    dst[idx] = (_Float16)src[(((size_t)a * S_ + j) * V_ + b) * H_ + h];
}
// straight f32 -> f16 copy (ov_w)
__global__ __launch_bounds__(256)
void pack_w_kernel(const float* __restrict__ src, _Float16* __restrict__ dst) {
    size_t idx = (size_t)blockIdx.x * 256 + threadIdx.x;   // over D*D
    dst[idx] = (_Float16)src[idx];
}

// ---------------------------------------------------------------------------
// Kernel 1: flash attention per (batch a, head h). Wave owns 16 query rows,
// block of 4 waves covers 64 rows. 64-key j-tiles: 16 WMMAs + 32 global + 4 LDS b128 loads
// per iteration, softmax bookkeeping amortized over 4 score tiles.
// att (f16) written as [B*S, D], d = feat*H + h.
// ---------------------------------------------------------------------------
__global__ __launch_bounds__(128)
void mha_flash_kernel(const _Float16* __restrict__ qp, const _Float16* __restrict__ kp,
                      const _Float16* __restrict__ vt, _Float16* __restrict__ att) {
    const int lane = threadIdx.x & 31;
    const int wave = threadIdx.x >> 5;
    const int g    = lane >> 4;
    const int ln   = lane & 15;
    const int a    = blockIdx.z;
    const int h    = blockIdx.y;
    const int i0   = blockIdx.x * 64 + wave * 16;

    __shared__ _Float16 ldsP[4][16][64];            // per-wave P restage tile (2KB)

    const size_t plane = ((size_t)a * H_ + h) * S_ * V_;   // qp/kp/vt head plane

    // Q fragments (already scaled by 1/sqrt(V) at pack time)
    const _Float16* qrow = qp + plane + (size_t)(i0 + ln) * V_;
    const v16h qa0 = ld16c(qrow + 16 * g);          // feats  0..31
    const v16h qa1 = ld16c(qrow + 32 + 16 * g);     // feats 32..63

    float mr[8], lr[8];
    v8f oacc[4];
#pragma unroll
    for (int r = 0; r < 8; ++r) { mr[r] = -1.0e30f; lr[r] = 0.0f; }
#pragma unroll
    for (int f = 0; f < 4; ++f) oacc[f] = (v8f){};

    for (int jt = 0; jt < S_; jt += 64) {
        // ---- scores: four 16x16 C tiles (keys jt+16*q .. jt+16*q+15) ----
        v8f c[4];
#pragma unroll
        for (int q = 0; q < 4; ++q) {
            const _Float16* krow = kp + plane + (size_t)(jt + 16 * q + ln) * V_;
            v8f t = wmma_f16(qa0, ld16c(krow + 16 * g), (v8f){});
            c[q]  = wmma_f16(qa1, ld16c(krow + 32 + 16 * g), t);
        }

        // prefetch next K tile (global_prefetch; inputs are L2-resident)
        if (jt + 64 < S_) {
            __builtin_prefetch(kp + plane + (size_t)(jt + 64 + ln) * V_, 0, 0);
        }

        // ---- online softmax update (rows M = r + 8*g) ----
#pragma unroll
        for (int r = 0; r < 8; ++r) {
            float t  = fmaxf(fmaxf(c[0][r], c[1][r]), fmaxf(c[2][r], c[3][r]));
            t        = redmax16(t);
            float nm = fmaxf(mr[r], t);
            float al = __expf(mr[r] - nm);
            float p0 = __expf(c[0][r] - nm);
            float p1 = __expf(c[1][r] - nm);
            float p2 = __expf(c[2][r] - nm);
            float p3 = __expf(c[3][r] - nm);
            float rs = redsum16((p0 + p1) + (p2 + p3));
            lr[r] = lr[r] * al + rs;
            mr[r] = nm;
            oacc[0][r] *= al;
            oacc[1][r] *= al;
            oacc[2][r] *= al;
            oacc[3][r] *= al;
            const int row = r + 8 * g;
            ldsP[wave][row][ln]      = (_Float16)p0;
            ldsP[wave][row][ln + 16] = (_Float16)p1;
            ldsP[wave][row][ln + 32] = (_Float16)p2;
            ldsP[wave][row][ln + 48] = (_Float16)p3;
        }
        asm volatile("s_wait_dscnt 0" ::: "memory");   // order LDS write -> read

        // ---- P back as two A fragments (keys 0..31 / 32..63 of the tile) ----
        // (same-wave DS ops are serviced in order, so next iteration's stores
        //  cannot pass these loads)
        const v16h pa0 = ld16c(&ldsP[wave][ln][16 * g]);
        const v16h pa1 = ld16c(&ldsP[wave][ln][32 + 16 * g]);

        // ---- O += P * V (4 feature chunks of 16) ----
#pragma unroll
        for (int f = 0; f < 4; ++f) {
            const _Float16* vrow = vt + plane + (size_t)(f * 16 + ln) * S_ + jt + 16 * g;
            v8f t   = wmma_f16(pa0, ld16c(vrow), oacc[f]);
            oacc[f] = wmma_f16(pa1, ld16c(vrow + 32), t);
        }
    }

    // ---- normalize and store att as f16 [B*S, D], d = feat*H + h ----
#pragma unroll
    for (int f = 0; f < 4; ++f) {
        const int feat = f * 16 + ln;
#pragma unroll
        for (int r = 0; r < 8; ++r) {
            const int row = i0 + r + 8 * g;
            const float val = oacc[f][r] / lr[r];
            att[((size_t)a * S_ + row) * D_ + feat * H_ + h] = (_Float16)val;
        }
    }
}

// ---------------------------------------------------------------------------
// Kernel 2: output projection  out[r,n] = sum_d att[r,d]*W[n,d] + bias[n]
// att f16 [R=B*S, D], wp f16 [D,D] row-major. One 16x16 tile per wave;
// all fragments are b128 load pairs.
// ---------------------------------------------------------------------------
__global__ __launch_bounds__(256)
void out_proj_kernel(const _Float16* __restrict__ att, const _Float16* __restrict__ wp,
                     const float* __restrict__ bias, float* __restrict__ out) {
    const int lane = threadIdx.x & 31;
    const int wave = threadIdx.x >> 5;
    const int g    = lane >> 4;
    const int ln   = lane & 15;
    const int n0   = blockIdx.x * 16;
    const int m0   = blockIdx.y * 128 + wave * 16;

    const _Float16* arow = att + (size_t)(m0 + ln) * D_;
    const _Float16* wrow = wp  + (size_t)(n0 + ln) * D_;

    v8f c = (v8f){};
    for (int d0 = 0; d0 < D_; d0 += 32) {
        c = wmma_f16(ld16c(arow + d0 + 16 * g), ld16c(wrow + d0 + 16 * g), c);
    }
    const float bb = bias[n0 + ln];
#pragma unroll
    for (int r = 0; r < 8; ++r) {
        out[(size_t)(m0 + r + 8 * g) * D_ + n0 + ln] = c[r] + bb;
    }
}

// ---------------------------------------------------------------------------
extern "C" void kernel_launch(void* const* d_in, const int* in_sizes, int n_in,
                              void* d_out, int out_size, void* d_ws, size_t ws_size,
                              hipStream_t stream) {
    const float* qv  = (const float*)d_in[0];
    const float* kv  = (const float*)d_in[1];
    const float* vv  = (const float*)d_in[2];
    const float* ovw = (const float*)d_in[3];
    const float* ovb = (const float*)d_in[4];
    float* out = (float*)d_out;

    // workspace layout (f16):  qp 8MB | kp 8MB | vt 8MB | wp 2MB | att 8MB
    char* ws = (char*)d_ws;
    _Float16* qp  = (_Float16*)(ws);
    _Float16* kp  = (_Float16*)(ws + 8u * 1024 * 1024);
    _Float16* vt  = (_Float16*)(ws + 16u * 1024 * 1024);
    _Float16* wp  = (_Float16*)(ws + 24u * 1024 * 1024);
    _Float16* att = (_Float16*)(ws + 26u * 1024 * 1024);

    const int nQKV = B_ * H_ * S_ * V_;                 // 4,194,304
    pack_headmajor_kernel<<<nQKV / 256, 256, 0, stream>>>(qv, qp, 0.125f);
    pack_headmajor_kernel<<<nQKV / 256, 256, 0, stream>>>(kv, kp, 1.0f);
    pack_v_transposed_kernel<<<nQKV / 256, 256, 0, stream>>>(vv, vt);
    pack_w_kernel<<<(D_ * D_) / 256, 256, 0, stream>>>(ovw, wp);

    dim3 g1(S_ / 64, H_, B_);                           // 4 waves/block
    mha_flash_kernel<<<g1, dim3(128), 0, stream>>>(qp, kp, vt, att);

    dim3 g2(D_ / 16, (B_ * S_) / 128);                  // 8 waves/block
    out_proj_kernel<<<g2, dim3(256), 0, stream>>>(att, wp, ovb, out);
}
